// LSTM_13941463843485
// MI455X (gfx1250) — matrix operand order repaired and checked by reading, not compiled, not run on hardware
//
#include <hip/hip_runtime.h>
#include <hip/hip_bf16.h>

typedef __attribute__((ext_vector_type(16))) _Float16 v16h;
typedef __attribute__((ext_vector_type(8)))  _Float16 v8h;
typedef __attribute__((ext_vector_type(4)))  _Float16 v4h;
typedef __attribute__((ext_vector_type(8)))  float    v8f;
typedef __attribute__((ext_vector_type(4)))  float    v4f;

#define DEV __device__ __forceinline__

namespace {
constexpr int IN   = 1260;   // layer-0 input width
constexpr int H    = 40;     // hidden
constexpr int G    = 160;    // 4*H gate width
constexpr int NOUT = 4;
constexpr int KC   = 64;                      // K-chunk staged in LDS (2 WMMA K-steps)
constexpr int NCH  = (IN + KC - 1) / KC;      // 20 chunks (last zero-padded)
constexpr int WST  = 72;                      // w_lds row stride (f16): 144B rows, 16B aligned
constexpr int GST  = 160;                     // gbuf row stride (f16): 320B rows, 16B aligned
constexpr int WAVES = 8;
constexpr int NTHR  = WAVES * 32;
constexpr int RPW   = 16;                     // rows (batch) per wave = WMMA M
constexpr int RPB   = WAVES * RPW;            // 128 rows per block
constexpr int QPT   = G * (KC / 4) / NTHR;    // 10 weight-quads staged per thread
// f-gate (columns 40..79) is dead since c0==0: tiles 3,4 (cols 48..79) skipped entirely.
constexpr int TILES[8] = {0, 1, 2, 5, 6, 7, 8, 9};
}

DEV float fsig(float x) { return 1.0f / (1.0f + __expf(-x)); }
DEV float ftanh(float x) {
  float e = __expf(-2.0f * fabsf(x));
  float r = (1.0f - e) / (1.0f + e);
  return (x >= 0.0f) ? r : -r;
}

DEV v16h pack16(v4f r0, v4f r1, v4f r2, v4f r3) {
  v16h a;
#pragma unroll
  for (int i = 0; i < 4; ++i) {
    a[i]      = (_Float16)r0[i];
    a[4 + i]  = (_Float16)r1[i];
    a[8 + i]  = (_Float16)r2[i];
    a[12 + i] = (_Float16)r3[i];
  }
  return a;
}

// Raw A-fragment gather (two 8-float runs: [kb..kb+7], [kb+16..kb+23]).
// 'fast' is UNIFORM per (chunk,kk): scalar branch, no exec-mask churn.
DEV void load_a_raw(const float* __restrict__ xrow, int kb, bool fast, v4f* r) {
  if (fast) {
    r[0] = *(const v4f*)(xrow + kb);
    r[1] = *(const v4f*)(xrow + kb + 4);
    r[2] = *(const v4f*)(xrow + kb + 16);
    r[3] = *(const v4f*)(xrow + kb + 20);
  } else {
#pragma unroll
    for (int j = 0; j < 4; ++j) {
      r[0][j] = (kb + j      < IN) ? xrow[kb + j]      : 0.0f;
      r[1][j] = (kb + 4 + j  < IN) ? xrow[kb + 4 + j]  : 0.0f;
      r[2][j] = (kb + 16 + j < IN) ? xrow[kb + 16 + j] : 0.0f;
      r[3][j] = (kb + 20 + j < IN) ? xrow[kb + 20 + j] : 0.0f;
    }
  }
}

// Issue global loads of the next 160x64 Wih0 slice into registers (f32).
DEV void stage_load(const float* __restrict__ W0, int k0c, int tid, v4f* wreg) {
#pragma unroll
  for (int i = 0; i < QPT; ++i) {
    const int q    = tid + i * NTHR;
    const int n    = q >> 4;
    const int kloc = (q & 15) << 2;
    const int kg   = k0c + kloc;
    if (kg + 3 < IN) {
      wreg[i] = *(const v4f*)(W0 + (size_t)n * IN + kg);
    } else {
#pragma unroll
      for (int j = 0; j < 4; ++j)
        wreg[i][j] = (kg + j < IN) ? W0[(size_t)n * IN + kg + j] : 0.0f;
    }
  }
}

// Convert staged registers to f16 and store into the (other) LDS buffer.
DEV void stage_store(_Float16* wbuf, int tid, const v4f* wreg) {
#pragma unroll
  for (int i = 0; i < QPT; ++i) {
    const int q    = tid + i * NTHR;
    const int n    = q >> 4;
    const int kloc = (q & 15) << 2;
    v4h o4;
#pragma unroll
    for (int j = 0; j < 4; ++j) o4[j] = (_Float16)wreg[i][j];
    *(v4h*)(wbuf + n * WST + kloc) = o4;
  }
}

// A-fragment from per-wave h buffer in LDS (K=40 zero-padded to 64).
DEV v16h load_a_h(const _Float16* hrow, int kk, int hi) {
  v16h a;
  if (kk == 0) {
    v8h r1 = *(const v8h*)(hrow + hi * 8);
    v8h r2 = *(const v8h*)(hrow + hi * 8 + 16);
    a = __builtin_shufflevector(r1, r2, 0,1,2,3,4,5,6,7,8,9,10,11,12,13,14,15);
  } else {
#pragma unroll
    for (int i = 0; i < 16; ++i) a[i] = (_Float16)0.0f;
    if (hi == 0) {                       // only K=32..39 are real
      v8h r1 = *(const v8h*)(hrow + 32);
#pragma unroll
      for (int i = 0; i < 8; ++i) a[i] = r1[i];
    }
  }
  return a;
}

// B-fragment (32x16) for the small layers, straight from global (L2-hot, 25.6KB).
DEV v16h load_b_small(const float* __restrict__ W, int n, int kk, int hi) {
  v16h b;
  if (kk == 0) {
    const v4f* p = (const v4f*)(W + n * H + hi * 8);
    return pack16(p[0], p[1], p[4], p[5]);
  }
#pragma unroll
  for (int i = 0; i < 16; ++i) b[i] = (_Float16)0.0f;
  if (hi == 0) {                         // K=32..39 valid, 40..47 zero
    v4f r0 = *(const v4f*)(W + n * H + 32);
    v4f r1 = *(const v4f*)(W + n * H + 36);
#pragma unroll
    for (int i = 0; i < 4; ++i) {
      b[i]     = (_Float16)r0[i];
      b[4 + i] = (_Float16)r1[i];
    }
  }
  return b;
}

// Store gate accumulators (+bias) to the per-wave LDS buffer, then apply the
// degenerate-LSTM activation in place: h = sig(o)*tanh(sig(i)*tanh(g)).
// All traffic is wave-local; LDS is in-order per wave, so a compiler fence
// (wave_barrier) is enough — no workgroup barrier needed.
DEV void gates_activate(_Float16* gwave, const v8f* acc,
                        const float* __restrict__ bih, const float* __restrict__ bhh,
                        int lane, int ln, int hi) {
#pragma unroll
  for (int t = 0; t < 8; ++t) {
    const int cn = TILES[t] * 16 + ln;
    const float bs = bih[cn] + bhh[cn];
#pragma unroll
    for (int r = 0; r < 8; ++r)          // D layout: lane half hi -> rows r + 8*hi
      gwave[(hi * 8 + r) * GST + cn] = (_Float16)(acc[t][r] + bs);
  }
  __builtin_amdgcn_wave_barrier();
#pragma unroll
  for (int p = 0; p < 20; ++p) {         // 16 rows * 40 h-values = 640 per wave
    const int idx = lane + 32 * p;
    const int row = idx / 40;
    const int j   = idx - row * 40;
    const float gi = (float)gwave[row * GST + j];
    const float gg = (float)gwave[row * GST + j + 80];
    const float go = (float)gwave[row * GST + j + 120];
    const float c  = fsig(gi) * ftanh(gg);
    gwave[row * GST + j] = (_Float16)(fsig(go) * ftanh(c));
  }
  __builtin_amdgcn_wave_barrier();
}

DEV void small_layer(_Float16* gwave, const float* __restrict__ W,
                     const float* __restrict__ bih, const float* __restrict__ bhh,
                     int lane, int ln, int hi) {
  const _Float16* hrow = gwave + ln * GST;   // A row M = ln for both lane halves
  v8f acc[8] = {};
#pragma unroll
  for (int kk = 0; kk < 2; ++kk) {
    v16h a = load_a_h(hrow, kk, hi);
    v16h bf[8];
#pragma unroll
    for (int t = 0; t < 8; ++t)
      bf[t] = load_b_small(W, TILES[t] * 16 + ln, kk, hi);
#pragma unroll
    for (int t = 0; t < 8; ++t)
      acc[t] = __builtin_amdgcn_wmma_f32_16x16x32_f16(
          false, a, false, bf[t], (short)0, acc[t], false, false);
  }
  gates_activate(gwave, acc, bih, bhh, lane, ln, hi);
}

__global__ __launch_bounds__(NTHR)
void lstm_fused_kernel(const float* __restrict__ x,
                       const float* __restrict__ Wih0, const float* __restrict__ bih0,
                       const float* __restrict__ bhh0,
                       const float* __restrict__ Wih1, const float* __restrict__ bih1,
                       const float* __restrict__ bhh1,
                       const float* __restrict__ Wih2, const float* __restrict__ bih2,
                       const float* __restrict__ bhh2,
                       const float* __restrict__ Wout, const float* __restrict__ bout,
                       float* __restrict__ out) {
  // Double-buffered Wih0 slice (2 x 23,040 B) + per-wave gates/h (40,960 B) = 87 KB
  // (WGP LDS is 320 KB; this also probes the >64 KB static-LDS path).
  __shared__ alignas(16) _Float16 wlds[2][G * WST];
  __shared__ alignas(16) _Float16 gbuf[WAVES * RPW * GST];

  const int tid  = threadIdx.x;
  const int lane = tid & 31;
  const int wv   = tid >> 5;
  const int ln   = lane & 15;
  const int hi   = lane >> 4;
  const int R0   = blockIdx.x * RPB;

  _Float16* gwave = gbuf + wv * RPW * GST;
  const float* xrow = x + (size_t)(R0 + wv * RPW + ln) * IN;

  // ---------------- Layer 0: gates = x @ Wih0^T, K = 1260 (padded to 1280) ----
  v4f wreg[QPT];
  stage_load(Wih0, 0, tid, wreg);
  stage_store(wlds[0], tid, wreg);
  __syncthreads();

  v8f acc[8] = {};
  for (int chunk = 0; chunk < NCH; ++chunk) {
    const int k0c = chunk * KC;
    _Float16* cur = wlds[chunk & 1];
    _Float16* nxt = wlds[(chunk & 1) ^ 1];
    const bool last = (chunk + 1 == NCH);

    // Issue this chunk's x loads FIRST (LOADcnt decrements in order, so the
    // A-fragment wait won't drag in the W staging loads issued after it).
    v4f ar[2][4];
    load_a_raw(xrow, k0c + hi * 8,      true,              ar[0]);
    load_a_raw(xrow, k0c + 32 + hi * 8, k0c + 64 <= IN,    ar[1]);
    if (!last) stage_load(Wih0, k0c + KC, tid, wreg);

    // 16 WMMAs while the global loads for chunk+1 are in flight.
#pragma unroll
    for (int kk = 0; kk < 2; ++kk) {
      v16h a = pack16(ar[kk][0], ar[kk][1], ar[kk][2], ar[kk][3]);
      v16h bf[8];
#pragma unroll
      for (int t = 0; t < 8; ++t) {
        const int cn = TILES[t] * 16 + ln;
        const v8h* bp = (const v8h*)(cur + cn * WST + kk * 32 + hi * 8);
        bf[t] = __builtin_shufflevector(bp[0], bp[2],
                                        0,1,2,3,4,5,6,7,8,9,10,11,12,13,14,15);
      }
#pragma unroll
      for (int t = 0; t < 8; ++t)
        acc[t] = __builtin_amdgcn_wmma_f32_16x16x32_f16(
            false, a, false, bf[t], (short)0, acc[t], false, false);
    }

    if (!last) stage_store(nxt, tid, wreg);
    __syncthreads();   // staging done / buffer free for next overwrite
  }
  gates_activate(gwave, acc, bih0, bhh0, lane, ln, hi);

  // ---------------- Layers 1 & 2 (K = 40, WMMA with zero-padded K) -----------
  small_layer(gwave, Wih1, bih1, bhh1, lane, ln, hi);
  small_layer(gwave, Wih2, bih2, bhh2, lane, ln, hi);

  // ---------------- Output linear: [16x40] @ [40x4] + b ----------------------
#pragma unroll
  for (int p = 0; p < 2; ++p) {
    const int idx = lane + 32 * p;     // 64 outputs per wave
    const int row = idx >> 2;
    const int o   = idx & 3;
    float s = bout[o];
#pragma unroll 8
    for (int j = 0; j < H; ++j)
      s += (float)gwave[row * GST + j] * Wout[o * H + j];
    out[(size_t)(R0 + wv * RPW + row) * NOUT + o] = s;
  }
}

extern "C" void kernel_launch(void* const* d_in, const int* in_sizes, int n_in,
                              void* d_out, int out_size, void* d_ws, size_t ws_size,
                              hipStream_t stream) {
  (void)in_sizes; (void)n_in; (void)out_size; (void)d_ws; (void)ws_size;
  const float* x    = (const float*)d_in[0];
  const float* Wih0 = (const float*)d_in[1];
  const float* bih0 = (const float*)d_in[3];
  const float* bhh0 = (const float*)d_in[4];
  const float* Wih1 = (const float*)d_in[5];
  const float* bih1 = (const float*)d_in[7];
  const float* bhh1 = (const float*)d_in[8];
  const float* Wih2 = (const float*)d_in[9];
  const float* bih2 = (const float*)d_in[11];
  const float* bhh2 = (const float*)d_in[12];
  const float* Wout = (const float*)d_in[13];
  const float* bout = (const float*)d_in[14];
  float* out = (float*)d_out;

  const int BATCH = 65536;
  dim3 grid(BATCH / RPB);   // 512 blocks of 128 rows
  dim3 block(NTHR);         // 8 wave32
  lstm_fused_kernel<<<grid, block, 0, stream>>>(
      x, Wih0, bih0, bhh0, Wih1, bih1, bhh1, Wih2, bih2, bhh2, Wout, bout, out);
}
